// GAU_54245436948590
// MI455X (gfx1250) — compile-verified, matrix operand order checked by reading, not run on hardware
//
#include <hip/hip_runtime.h>
#include <hip/hip_bf16.h>

#define DI __device__ __forceinline__

typedef __bf16 bf16_t;
typedef __attribute__((ext_vector_type(16))) __bf16 bf16x16;
typedef __attribute__((ext_vector_type(8)))  float  f32x8;
typedef unsigned int u32x4 __attribute__((ext_vector_type(4)));
typedef int i32x4 __attribute__((ext_vector_type(4)));
typedef int i32x8 __attribute__((ext_vector_type(8)));

constexpr int Bz = 4, Lz = 4096, Hz = 512, Ez = 1024, Sz = 128;
constexpr int Mz  = Bz * Lz;          // 16384 rows total
constexpr int NUV = 2 * Ez + Sz;      // 2176
constexpr float INV_SQRT_S = 0.08838834764831845f; // 1/sqrt(128)

#if __has_builtin(__builtin_amdgcn_tensor_load_to_lds)
#define GAU_HAVE_TDM 1
#else
#define GAU_HAVE_TDM 0
#endif

// ---------------- helpers ----------------

DI bf16_t f2bf(float f) {
  unsigned u = __builtin_bit_cast(unsigned, f);
  unsigned r = (u + 0x7FFFu + ((u >> 16) & 1u)) >> 16;
  return __builtin_bit_cast(bf16_t, (unsigned short)r);
}

DI f32x8 fzero() {
  f32x8 z;
#pragma unroll
  for (int i = 0; i < 8; ++i) z[i] = 0.0f;
  return z;
}

DI float silu_fast(float x) {
  // x * sigmoid(x) with fast v_rcp_f32 (no IEEE division chain)
  return x * __builtin_amdgcn_rcpf(1.0f + __expf(-x));
}

// Load a 16x32 bf16 WMMA fragment from "n-major" storage:
// 16 rows (one per lane-id mod 16), 32 contiguous K elements per row.
// Per ISA layout: lanes 0-15 hold K=[0..7]+[16..23], lanes 16-31 hold K=[8..15]+[24..31].
DI bf16x16 frag_nmajor(const bf16_t* base, int ld, int lane) {
  const int r  = lane & 15;
  const int kh = (lane >> 4) & 1;
  const bf16_t* p = base + (size_t)r * ld + kh * 8;
  bf16x16 f;
#pragma unroll
  for (int i = 0; i < 8; ++i) f[i] = p[i];
#pragma unroll
  for (int i = 0; i < 8; ++i) f[8 + i] = p[16 + i];
  return f;
}

DI f32x8 wmma_bf16(bf16x16 a, bf16x16 b, f32x8 c) {
  return __builtin_amdgcn_wmma_f32_16x16x32_bf16(false, a, false, b,
                                                 (short)0, c, false, false);
}

#if GAU_HAVE_TDM
// Tensor Data Mover: DMA a 2-D bf16 tile (tile_w elems x tile_h rows) from a
// row-major tensor (row stride = stride_elems) into LDS at lds_byte_off.
// D# layout per CDNA5 ISA (group0: count/lds/global/type,
// group1: data_size=2B, dims/strides/tile dims). Issued once per wave.
DI void tdm_load_bf16_tile(const bf16_t* gptr, unsigned lds_byte_off,
                           int tile_w, int tile_h,
                           int tensor_w, int tensor_h, int stride_elems) {
  const unsigned long long ga = (unsigned long long)(uintptr_t)gptr;
  u32x4 g0;
  g0[0] = 1u;                                   // count=1 (valid), user mode
  g0[1] = lds_byte_off;                         // lds_addr
  g0[2] = (unsigned)ga;                         // global_addr[31:0]
  g0[3] = (unsigned)((ga >> 32) & 0x01FFFFFFu)  // global_addr[56:32]
          | (2u << 30);                         // type=2 ("image")
  i32x8 g1;
  g1[0] = (int)(1u << 16);                      // wg_mask=0, data_size=1 (2B)
  g1[1] = (int)(((unsigned)tensor_w & 0xFFFFu) << 16);            // dim0 lo16
  g1[2] = (int)((((unsigned)tensor_w >> 16) & 0xFFFFu)            // dim0 hi16
                | (((unsigned)tensor_h & 0xFFFFu) << 16));        // dim1 lo16
  g1[3] = (int)((((unsigned)tensor_h >> 16) & 0xFFFFu)            // dim1 hi16
                | (((unsigned)tile_w & 0xFFFFu) << 16));          // tile_dim0
  g1[4] = (int)((unsigned)tile_h & 0xFFFFu);                      // tile_dim1
  g1[5] = (int)(unsigned)stride_elems;          // tensor_dim0_stride[31:0]
  g1[6] = 0;
  g1[7] = 0;
  i32x4 gz;
  gz[0] = 0; gz[1] = 0; gz[2] = 0; gz[3] = 0;
#if defined(__clang_major__) && (__clang_major__ >= 23)
  i32x8 gz8;
#pragma unroll
  for (int i = 0; i < 8; ++i) gz8[i] = 0;
  __builtin_amdgcn_tensor_load_to_lds(g0, g1, gz, gz, gz8, 0);
#else
  __builtin_amdgcn_tensor_load_to_lds(g0, g1, gz, gz, 0);
#endif
}
#endif  // GAU_HAVE_TDM

DI unsigned lds_offset_of(const void* p) {
  // generic shared pointer = {aperture_hi, lds_offset}; low 32 bits = offset
  return (unsigned)(uintptr_t)p;
}

// ---------------- kernel 1: LayerNorm -> bf16 ----------------

__global__ __launch_bounds__(128) void GAU_ln(const float* __restrict__ x,
                                              const float* __restrict__ w,
                                              const float* __restrict__ b,
                                              bf16_t* __restrict__ xn) {
  __shared__ float r1[128], r2[128];
  const int row = blockIdx.x, tid = threadIdx.x;
  const float* xr = x + (size_t)row * Hz;
  float vals[4];
  float s = 0.0f, s2 = 0.0f;
#pragma unroll
  for (int i = 0; i < 4; ++i) {
    vals[i] = xr[tid + i * 128];
    s += vals[i];
    s2 += vals[i] * vals[i];
  }
  r1[tid] = s; r2[tid] = s2;
  __syncthreads();
  for (int off = 64; off > 0; off >>= 1) {
    if (tid < off) { r1[tid] += r1[tid + off]; r2[tid] += r2[tid + off]; }
    __syncthreads();
  }
  const float mu  = r1[0] * (1.0f / Hz);
  const float var = r2[0] * (1.0f / Hz) - mu * mu;
  const float rstd = rsqrtf(var + 1e-5f);
#pragma unroll
  for (int i = 0; i < 4; ++i) {
    const int c = tid + i * 128;
    xn[(size_t)row * Hz + c] = f2bf((vals[i] - mu) * rstd * w[c] + b[c]);
  }
}

// ---------------- kernel 2: weights fp32 -> bf16 ----------------

__global__ void GAU_cvt(const float* __restrict__ uvw, const float* __restrict__ ow,
                        bf16_t* __restrict__ wuv, bf16_t* __restrict__ wo) {
  const int i = blockIdx.x * blockDim.x + threadIdx.x;
  const int n1 = NUV * Hz;
  const int n2 = Hz * Ez;
  if (i < n1) wuv[i] = f2bf(uvw[i]);
  else if (i < n1 + n2) wo[i - n1] = f2bf(ow[i - n1]);
}

// ---------------- kernel 3: uv = silu(xn @ W^T + b) ----------------
// 64(M) x 128(N) tile per block; 8 waves arranged 4x2; K-loop of 32.
// Double-buffered LDS; both tiles DMA'd by the Tensor Data Mover (wave 0)
// one step ahead of compute; s_wait_tensorcnt placed after the WMMAs so the
// DMA overlaps the matrix math. Fallback: cooperative vector loads.

__global__ __launch_bounds__(256) void GAU_uvgemm(const bf16_t* __restrict__ xn,
                                                  const bf16_t* __restrict__ wuv,
                                                  const float* __restrict__ bias,
                                                  bf16_t* __restrict__ u,
                                                  bf16_t* __restrict__ v,
                                                  float* __restrict__ basef) {
  __shared__ __align__(16) bf16_t sA[2][64 * 32];
  __shared__ __align__(16) bf16_t sB[2][128 * 32];
  const int m0 = blockIdx.x * 64;
  const int n0 = blockIdx.y * 128;
  const int tid = threadIdx.x;
  const int wid = tid >> 5, lane = tid & 31;
  const int wm = wid >> 1, wn = wid & 1;
  const int NK = Hz / 32;

  f32x8 acc[4];
#pragma unroll
  for (int j = 0; j < 4; ++j) acc[j] = fzero();

  // ---- prologue: stage tile 0 into buffer 0 ----
#if GAU_HAVE_TDM
  if (wid == 0) {
    tdm_load_bf16_tile(xn + (size_t)m0 * Hz, lds_offset_of(&sA[0][0]),
                       32, 64, Hz, Mz, Hz);
    tdm_load_bf16_tile(wuv + (size_t)n0 * Hz, lds_offset_of(&sB[0][0]),
                       32, 128, Hz, NUV, Hz);
    __builtin_amdgcn_s_wait_tensorcnt(0);
  }
#else
  {
    const int row = tid >> 2, c8 = (tid & 3) * 8;
    *(uint4*)&sA[0][row * 32 + c8] =
        *(const uint4*)&xn[(size_t)(m0 + row) * Hz + c8];
    const int rowb = tid >> 1, c16 = (tid & 1) * 16;
    const uint4* src = (const uint4*)&wuv[(size_t)(n0 + rowb) * Hz + c16];
    uint4* dst = (uint4*)&sB[0][rowb * 32 + c16];
    dst[0] = src[0];
    dst[1] = src[1];
  }
#endif
  __syncthreads();

  for (int i = 0; i < NK; ++i) {
    const int cur = i & 1, nxt = cur ^ 1;
    const int k1 = (i + 1) * 32;
    // stage next tile while computing the current one
    if (i + 1 < NK) {
#if GAU_HAVE_TDM
      if (wid == 0) {
        tdm_load_bf16_tile(xn + (size_t)m0 * Hz + k1, lds_offset_of(&sA[nxt][0]),
                           32, 64, Hz, Mz, Hz);
        tdm_load_bf16_tile(wuv + (size_t)n0 * Hz + k1, lds_offset_of(&sB[nxt][0]),
                           32, 128, Hz, NUV, Hz);
      }
#else
      const int row = tid >> 2, c8 = (tid & 3) * 8;
      *(uint4*)&sA[nxt][row * 32 + c8] =
          *(const uint4*)&xn[(size_t)(m0 + row) * Hz + k1 + c8];
      const int rowb = tid >> 1, c16 = (tid & 1) * 16;
      const uint4* src = (const uint4*)&wuv[(size_t)(n0 + rowb) * Hz + k1 + c16];
      uint4* dst = (uint4*)&sB[nxt][rowb * 32 + c16];
      dst[0] = src[0];
      dst[1] = src[1];
#endif
    }
    const bf16x16 af = frag_nmajor(&sA[cur][(wm * 16) * 32], 32, lane);
#pragma unroll
    for (int j = 0; j < 4; ++j) {
      const bf16x16 bf = frag_nmajor(&sB[cur][(wn * 64 + j * 16) * 32], 32, lane);
      acc[j] = wmma_bf16(af, bf, acc[j]);
    }
#if GAU_HAVE_TDM
    if (wid == 0 && i + 1 < NK) __builtin_amdgcn_s_wait_tensorcnt(0);
#endif
    __syncthreads();
  }

  const int ncol = lane & 15, rbase = (lane >> 4) * 8;
#pragma unroll
  for (int j = 0; j < 4; ++j) {
#pragma unroll
    for (int r = 0; r < 8; ++r) {
      const int ng = n0 + wn * 64 + j * 16 + ncol;
      const int mg = m0 + wm * 16 + rbase + r;
      const float val = silu_fast(acc[j][r] + bias[ng]);
      if (ng < Ez)            u[(size_t)mg * Ez + ng]            = f2bf(val);
      else if (ng < 2 * Ez)   v[(size_t)mg * Ez + (ng - Ez)]     = f2bf(val);
      else                    basef[(size_t)mg * Sz + (ng - 2 * Ez)] = val;
    }
  }
}

// ---------------- kernel 4: gamma/beta + RoPE -> q,k (bf16) ----------------

__global__ __launch_bounds__(64) void GAU_rope(const float* __restrict__ basef,
                                               const float* __restrict__ gamma,
                                               const float* __restrict__ beta,
                                               bf16_t* __restrict__ qb,
                                               bf16_t* __restrict__ kb) {
  const int row = blockIdx.x;
  const int s = threadIdx.x; // 0..63
  const float pos = (float)(row & (Lz - 1));
  const float freq = __powf(10000.0f, (float)s * (1.0f / 64.0f));
  float sn, cs;
  __sincosf(pos * freq, &sn, &cs);
  const float b0 = basef[(size_t)row * Sz + s];
  const float b1 = basef[(size_t)row * Sz + 64 + s];
  { // head 0 -> q
    const float x1 = b0 * gamma[s] + beta[s];
    const float x2 = b1 * gamma[64 + s] + beta[64 + s];
    qb[(size_t)row * Sz + s]      = f2bf(x1 * cs - x2 * sn);
    qb[(size_t)row * Sz + 64 + s] = f2bf(x2 * cs + x1 * sn);
  }
  { // head 1 -> k
    const float x1 = b0 * gamma[Sz + s] + beta[Sz + s];
    const float x2 = b1 * gamma[Sz + 64 + s] + beta[Sz + 64 + s];
    kb[(size_t)row * Sz + s]      = f2bf(x1 * cs - x2 * sn);
    kb[(size_t)row * Sz + 64 + s] = f2bf(x2 * cs + x1 * sn);
  }
}

// ---------------- kernel 5: v (MxE) -> vT (ExM) ----------------

__global__ __launch_bounds__(256) void GAU_transpose(const bf16_t* __restrict__ v,
                                                     bf16_t* __restrict__ vT) {
  __shared__ bf16_t t[32][33];
  const int tx = threadIdx.x, ty = threadIdx.y;
  const int e0 = blockIdx.x * 32, m0 = blockIdx.y * 32;
#pragma unroll
  for (int kk = 0; kk < 4; ++kk)
    t[ty + kk * 8][tx] = v[(size_t)(m0 + ty + kk * 8) * Ez + e0 + tx];
  __syncthreads();
#pragma unroll
  for (int kk = 0; kk < 4; ++kk)
    vT[(size_t)(e0 + ty + kk * 8) * Mz + m0 + tx] = t[tx][ty + kk * 8];
}

// ---------------- kernel 6: flash-style quadratic attention ----------------
// block: 16 query rows x 8 waves. Pipelined: wave 0 computes the relu^2
// score tile for key-tile t+1 (WMMA q.kT) into the spare LDS buffer while
// ALL waves (incl. wave 0) consume tile t's scores as an A fragment and
// accumulate kernel @ v over their 128-wide E slice. One barrier per tile.

__global__ __launch_bounds__(256) void GAU_attn(const bf16_t* __restrict__ qb,
                                                const bf16_t* __restrict__ kb,
                                                const bf16_t* __restrict__ vT,
                                                const bf16_t* __restrict__ ub,
                                                bf16_t* __restrict__ gb) {
  __shared__ bf16_t sS[2][16 * 32];
  const int b = blockIdx.y;
  const int l0 = blockIdx.x * 16;
  const int tid = threadIdx.x;
  const int wid = tid >> 5, lane = tid & 31;
  const int rowq = b * Lz + l0;
  const int e0 = wid * 128;
  const int NT = Lz / 32; // 128 key tiles

  bf16x16 qf[4];
  if (wid == 0) {
#pragma unroll
    for (int ks = 0; ks < 4; ++ks)
      qf[ks] = frag_nmajor(qb + (size_t)rowq * Sz + ks * 32, Sz, lane);
  }

  f32x8 acc[8];
#pragma unroll
  for (int j = 0; j < 8; ++j) acc[j] = fzero();

  // ---- prologue: scores for key tile 0 -> sS[0] ----
  if (wid == 0) {
    f32x8 c0 = fzero(), c1 = fzero();
#pragma unroll
    for (int ks = 0; ks < 4; ++ks) {
      c0 = wmma_bf16(qf[ks],
                     frag_nmajor(kb + (size_t)(b * Lz) * Sz + ks * 32, Sz, lane), c0);
      c1 = wmma_bf16(qf[ks],
                     frag_nmajor(kb + (size_t)(b * Lz + 16) * Sz + ks * 32, Sz, lane), c1);
    }
    const int ncol = lane & 15, rbase = (lane >> 4) * 8;
#pragma unroll
    for (int r = 0; r < 8; ++r) {
      float a = fmaxf(c0[r] * INV_SQRT_S, 0.0f);
      float d = fmaxf(c1[r] * INV_SQRT_S, 0.0f);
      sS[0][(rbase + r) * 32 + ncol]      = f2bf(a * a);
      sS[0][(rbase + r) * 32 + 16 + ncol] = f2bf(d * d);
    }
  }
  __syncthreads();

  for (int it = 0; it < NT; ++it) {
    const int cur = it & 1, nxt = cur ^ 1;
    const int m0 = it * 32;
    if (wid == 0 && it + 1 < NT) { // scores for next tile while others compute
      const int m1 = m0 + 32;
      __builtin_prefetch(kb + (size_t)(b * Lz + m1) * Sz, 0, 3);
      f32x8 c0 = fzero(), c1 = fzero();
#pragma unroll
      for (int ks = 0; ks < 4; ++ks) {
        c0 = wmma_bf16(qf[ks],
                       frag_nmajor(kb + (size_t)(b * Lz + m1) * Sz + ks * 32, Sz, lane), c0);
        c1 = wmma_bf16(qf[ks],
                       frag_nmajor(kb + (size_t)(b * Lz + m1 + 16) * Sz + ks * 32, Sz, lane), c1);
      }
      const int ncol = lane & 15, rbase = (lane >> 4) * 8;
#pragma unroll
      for (int r = 0; r < 8; ++r) {
        float a = fmaxf(c0[r] * INV_SQRT_S, 0.0f);
        float d = fmaxf(c1[r] * INV_SQRT_S, 0.0f);
        sS[nxt][(rbase + r) * 32 + ncol]      = f2bf(a * a);
        sS[nxt][(rbase + r) * 32 + 16 + ncol] = f2bf(d * d);
      }
    }
    const bf16x16 af = frag_nmajor(&sS[cur][0], 32, lane);
#pragma unroll
    for (int j = 0; j < 8; ++j) {
      const bf16x16 bv = frag_nmajor(
          vT + (size_t)(e0 + j * 16) * Mz + (b * Lz + m0), Mz, lane);
      acc[j] = wmma_bf16(af, bv, acc[j]);
    }
    __syncthreads();
  }

  const int ncol = lane & 15, rbase = (lane >> 4) * 8;
#pragma unroll
  for (int j = 0; j < 8; ++j) {
#pragma unroll
    for (int r = 0; r < 8; ++r) {
      const int lrow = l0 + rbase + r;
      const int col = e0 + j * 16 + ncol;
      const size_t idx = (size_t)(b * Lz + lrow) * Ez + col;
      gb[idx] = f2bf((float)ub[idx] * acc[j][r]);
    }
  }
}

// ---------------- kernel 7: out = gated @ o_w^T + o_b + shortcut ----------------

__global__ __launch_bounds__(256) void GAU_ogemm(const bf16_t* __restrict__ g,
                                                 const bf16_t* __restrict__ wo,
                                                 const float* __restrict__ ob,
                                                 const float* __restrict__ x,
                                                 float* __restrict__ out) {
  __shared__ __align__(16) bf16_t sA[2][64 * 32];
  __shared__ __align__(16) bf16_t sB[2][128 * 32];
  const int m0 = blockIdx.x * 64;
  const int n0 = blockIdx.y * 128;
  const int tid = threadIdx.x;
  const int wid = tid >> 5, lane = tid & 31;
  const int wm = wid >> 1, wn = wid & 1;
  const int NK = Ez / 32;

  f32x8 acc[4];
#pragma unroll
  for (int j = 0; j < 4; ++j) acc[j] = fzero();

#if GAU_HAVE_TDM
  if (wid == 0) {
    tdm_load_bf16_tile(g + (size_t)m0 * Ez, lds_offset_of(&sA[0][0]),
                       32, 64, Ez, Mz, Ez);
    tdm_load_bf16_tile(wo + (size_t)n0 * Ez, lds_offset_of(&sB[0][0]),
                       32, 128, Ez, Hz, Ez);
    __builtin_amdgcn_s_wait_tensorcnt(0);
  }
#else
  {
    const int row = tid >> 2, c8 = (tid & 3) * 8;
    *(uint4*)&sA[0][row * 32 + c8] =
        *(const uint4*)&g[(size_t)(m0 + row) * Ez + c8];
    const int rowb = tid >> 1, c16 = (tid & 1) * 16;
    const uint4* src = (const uint4*)&wo[(size_t)(n0 + rowb) * Ez + c16];
    uint4* dst = (uint4*)&sB[0][rowb * 32 + c16];
    dst[0] = src[0];
    dst[1] = src[1];
  }
#endif
  __syncthreads();

  for (int i = 0; i < NK; ++i) {
    const int cur = i & 1, nxt = cur ^ 1;
    const int k1 = (i + 1) * 32;
    if (i + 1 < NK) {
#if GAU_HAVE_TDM
      if (wid == 0) {
        tdm_load_bf16_tile(g + (size_t)m0 * Ez + k1, lds_offset_of(&sA[nxt][0]),
                           32, 64, Ez, Mz, Ez);
        tdm_load_bf16_tile(wo + (size_t)n0 * Ez + k1, lds_offset_of(&sB[nxt][0]),
                           32, 128, Ez, Hz, Ez);
      }
#else
      const int row = tid >> 2, c8 = (tid & 3) * 8;
      *(uint4*)&sA[nxt][row * 32 + c8] =
          *(const uint4*)&g[(size_t)(m0 + row) * Ez + k1 + c8];
      const int rowb = tid >> 1, c16 = (tid & 1) * 16;
      const uint4* src = (const uint4*)&wo[(size_t)(n0 + rowb) * Ez + k1 + c16];
      uint4* dst = (uint4*)&sB[nxt][rowb * 32 + c16];
      dst[0] = src[0];
      dst[1] = src[1];
#endif
    }
    const bf16x16 af = frag_nmajor(&sA[cur][(wm * 16) * 32], 32, lane);
#pragma unroll
    for (int j = 0; j < 4; ++j) {
      const bf16x16 bf = frag_nmajor(&sB[cur][(wn * 64 + j * 16) * 32], 32, lane);
      acc[j] = wmma_bf16(af, bf, acc[j]);
    }
#if GAU_HAVE_TDM
    if (wid == 0 && i + 1 < NK) __builtin_amdgcn_s_wait_tensorcnt(0);
#endif
    __syncthreads();
  }

  const int ncol = lane & 15, rbase = (lane >> 4) * 8;
#pragma unroll
  for (int j = 0; j < 4; ++j) {
#pragma unroll
    for (int r = 0; r < 8; ++r) {
      const int ng = n0 + wn * 64 + j * 16 + ncol;
      const int mg = m0 + wm * 16 + rbase + r;
      out[(size_t)mg * Hz + ng] =
          acc[j][r] + ob[ng] + x[(size_t)mg * Hz + ng];
    }
  }
}

// ---------------- workspace layout (bytes) ----------------

constexpr size_t OFF_XN   = 0;                                    // M*H bf16
constexpr size_t OFF_WUV  = OFF_XN   + (size_t)Mz * Hz * 2;       // NUV*H bf16
constexpr size_t OFF_WO   = OFF_WUV  + (size_t)NUV * Hz * 2;      // H*E bf16
constexpr size_t OFF_U    = OFF_WO   + (size_t)Hz * Ez * 2;       // M*E bf16
constexpr size_t OFF_V    = OFF_U    + (size_t)Mz * Ez * 2;       // M*E bf16
constexpr size_t OFF_VT   = OFF_V    + (size_t)Mz * Ez * 2;       // E*M bf16
constexpr size_t OFF_BASE = OFF_VT   + (size_t)Mz * Ez * 2;       // M*S f32
constexpr size_t OFF_Q    = OFF_BASE + (size_t)Mz * Sz * 4;       // M*S bf16
constexpr size_t OFF_K    = OFF_Q    + (size_t)Mz * Sz * 2;       // M*S bf16
constexpr size_t OFF_G    = OFF_V;  // alias: v is dead after transpose

extern "C" void kernel_launch(void* const* d_in, const int* in_sizes, int n_in,
                              void* d_out, int out_size, void* d_ws, size_t ws_size,
                              hipStream_t stream) {
  (void)in_sizes; (void)n_in; (void)out_size; (void)ws_size;
  const float* x     = (const float*)d_in[0];
  const float* ln_w  = (const float*)d_in[1];
  const float* ln_b  = (const float*)d_in[2];
  const float* uv_w  = (const float*)d_in[3];
  const float* uv_b  = (const float*)d_in[4];
  const float* gamma = (const float*)d_in[5];
  const float* beta  = (const float*)d_in[6];
  const float* o_w   = (const float*)d_in[7];
  const float* o_b   = (const float*)d_in[8];
  float* out = (float*)d_out;

  char* ws = (char*)d_ws;
  bf16_t* xn    = (bf16_t*)(ws + OFF_XN);
  bf16_t* wuv   = (bf16_t*)(ws + OFF_WUV);
  bf16_t* wo    = (bf16_t*)(ws + OFF_WO);
  bf16_t* u     = (bf16_t*)(ws + OFF_U);
  bf16_t* v     = (bf16_t*)(ws + OFF_V);
  bf16_t* vT    = (bf16_t*)(ws + OFF_VT);
  float*  basef = (float*) (ws + OFF_BASE);
  bf16_t* qb    = (bf16_t*)(ws + OFF_Q);
  bf16_t* kb    = (bf16_t*)(ws + OFF_K);
  bf16_t* g     = (bf16_t*)(ws + OFF_G);

  GAU_ln<<<Mz, 128, 0, stream>>>(x, ln_w, ln_b, xn);

  const int ncvt = NUV * Hz + Hz * Ez;
  GAU_cvt<<<(ncvt + 255) / 256, 256, 0, stream>>>(uv_w, o_w, wuv, wo);

  GAU_uvgemm<<<dim3(Mz / 64, NUV / 128), 256, 0, stream>>>(xn, wuv, uv_b, u, v, basef);

  GAU_rope<<<Mz, 64, 0, stream>>>(basef, gamma, beta, qb, kb);

  GAU_transpose<<<dim3(Ez / 32, Mz / 32), dim3(32, 8), 0, stream>>>(v, vT);

  GAU_attn<<<dim3(Lz / 16, Bz), 256, 0, stream>>>(qb, kb, vT, u, g);

  GAU_ogemm<<<dim3(Mz / 64, Hz / 128), 256, 0, stream>>>(g, wo, o_b, x, out);
}
